// DeformBlock_9912784519491
// MI455X (gfx1250) — matrix operand order repaired and checked by reading, not compiled, not run on hardware
//
#include <hip/hip_runtime.h>
#include <hip/hip_bf16.h>
#include <stdint.h>

// ---------------------------------------------------------------------------
// Problem constants
// ---------------------------------------------------------------------------
#define NCH   128           // NC
#define NVTX  150000        // NV (divisible by 16: 9375 tiles)
#define NFACE 300000        // NF
#define LVOL  192

typedef __attribute__((ext_vector_type(16))) _Float16 v16h;
typedef __attribute__((ext_vector_type(8)))  float    v8f;

union AFrag { uint32_t u[8]; v16h v; };

__device__ __forceinline__ int clampi(int x, int lo, int hi) {
    return x < lo ? lo : (x > hi ? hi : x);
}

// ---------------------------------------------------------------------------
// 0) zero a float buffer (normal accumulator)
// ---------------------------------------------------------------------------
__global__ void zero_f32(float* p, int n) {
    for (int i = blockIdx.x * blockDim.x + threadIdx.x; i < n;
         i += gridDim.x * blockDim.x)
        p[i] = 0.0f;
}

// ---------------------------------------------------------------------------
// 1) face-normal scatter (atomicAdd f32)
// ---------------------------------------------------------------------------
__global__ void face_normals(const float* __restrict__ v,
                             const int*   __restrict__ f,
                             float* __restrict__ nacc, int nf) {
    int i = blockIdx.x * blockDim.x + threadIdx.x;
    if (i >= nf) return;
    int a = f[i * 3 + 0], b = f[i * 3 + 1], c = f[i * 3 + 2];
    float p0x = v[a*3+0], p0y = v[a*3+1], p0z = v[a*3+2];
    float p1x = v[b*3+0], p1y = v[b*3+1], p1z = v[b*3+2];
    float p2x = v[c*3+0], p2y = v[c*3+1], p2z = v[c*3+2];
    float e1x = p1x - p0x, e1y = p1y - p0y, e1z = p1z - p0z;
    float e2x = p2x - p0x, e2y = p2y - p0y, e2z = p2z - p0z;
    float nx = e1y * e2z - e1z * e2y;
    float ny = e1z * e2x - e1x * e2z;
    float nz = e1x * e2y - e1y * e2x;
    atomicAdd(&nacc[a*3+0], nx); atomicAdd(&nacc[a*3+1], ny); atomicAdd(&nacc[a*3+2], nz);
    atomicAdd(&nacc[b*3+0], nx); atomicAdd(&nacc[b*3+1], ny); atomicAdd(&nacc[b*3+2], nz);
    atomicAdd(&nacc[c*3+0], nx); atomicAdd(&nacc[c*3+1], ny); atomicAdd(&nacc[c*3+2], nz);
}

// ---------------------------------------------------------------------------
// 2) 2x2x2 average pooling (d = output dim)
// ---------------------------------------------------------------------------
__global__ void pool2(const float* __restrict__ src, float* __restrict__ dst, int d) {
    int t = blockIdx.x * blockDim.x + threadIdx.x;
    int total = d * d * d;
    if (t >= total) return;
    int k = t % d, j = (t / d) % d, i = t / (d * d);
    int s = d * 2;
    float sum = 0.f;
    #pragma unroll
    for (int di = 0; di < 2; ++di)
        #pragma unroll
        for (int dj = 0; dj < 2; ++dj)
            #pragma unroll
            for (int dk = 0; dk < 2; ++dk)
                sum += src[((size_t)(2*i+di) * s + (2*j+dj)) * s + (2*k+dk)];
    dst[t] = sum * 0.125f;
}

// ---------------------------------------------------------------------------
// 3) fold conv_w (128 x 375) with lfc_w (128 x 128)
// ---------------------------------------------------------------------------
__global__ void combine_conv_lfc(const float* __restrict__ conv_w,
                                 const float* __restrict__ conv_b,
                                 const float* __restrict__ lfc_w,
                                 const float* __restrict__ lfc_b,
                                 float* __restrict__ Wcomb,
                                 float* __restrict__ bcomb) {
    int t = blockIdx.x * blockDim.x + threadIdx.x;
    if (t < 375 * 128) {
        int k = t / 128, j = t % 128;
        float s = 0.f;
        for (int o = 0; o < 128; ++o)
            s += conv_w[(size_t)o * 375 + k] * lfc_w[(size_t)o * 128 + j];
        Wcomb[(size_t)k * 128 + j] = s;
    }
    if (t < 128) {
        float s = lfc_b[t];
        for (int o = 0; o < 128; ++o)
            s += conv_b[o] * lfc_w[(size_t)o * 128 + t];
        bcomb[t] = s;
    }
}

// ---------------------------------------------------------------------------
// 4) pre-swizzle row-major f32 weight [Kdim x Ndim] into per-lane WMMA
//    B-fragment order (f16): lane=(h<<4)|n, e: K=ks*32+8h+e (e<8), K=ks*32+16+8h+(e-8) (e>=8), N=nt*16+n
// ---------------------------------------------------------------------------
__global__ void prep_bfrag(const float* __restrict__ W, _Float16* __restrict__ out,
                           int Kdim, int Ndim, int nk, int ntiles) {
    int t = blockIdx.x * blockDim.x + threadIdx.x;
    int total = ntiles * nk * 32;
    if (t >= total) return;
    int lane = t % 32;
    int ks   = (t / 32) % nk;
    int nt   = t / (32 * nk);
    int h = lane >> 4, nn = lane & 15;
    int N = nt * 16 + nn;
    _Float16* dst = out + (size_t)t * 16;
    #pragma unroll
    for (int e = 0; e < 16; ++e) {
        int K = ks * 32 + (e < 8 ? 8 * h + e : 16 + 8 * h + (e - 8));
        float val = (K < Kdim) ? W[(size_t)K * Ndim + N] : 0.0f;
        dst[e] = (_Float16)val;
    }
}

// ---------------------------------------------------------------------------
// 5) fc1: normalize normals, 6 -> 128, leaky(0.15), write f16 xbuf[:,0:128]
// ---------------------------------------------------------------------------
__global__ void fc1_kernel(const float* __restrict__ v,
                           const float* __restrict__ nacc,
                           const float* __restrict__ w,   // (6,128)
                           const float* __restrict__ b,
                           _Float16* __restrict__ xbuf, int nv) {
    int i = blockIdx.x * blockDim.x + threadIdx.x;
    if (i >= nv) return;
    float in[6];
    in[0] = v[i*3+0]; in[1] = v[i*3+1]; in[2] = v[i*3+2];
    float nx = nacc[i*3+0], ny = nacc[i*3+1], nz = nacc[i*3+2];
    float nrm = sqrtf(nx*nx + ny*ny + nz*nz);
    nrm = fmaxf(nrm, 1e-12f);
    in[3] = nx / nrm; in[4] = ny / nrm; in[5] = nz / nrm;
    _Float16* dst = xbuf + (size_t)i * 256;
    for (int j = 0; j < 128; ++j) {
        float s = b[j];
        #pragma unroll
        for (int k = 0; k < 6; ++k) s += in[k] * w[k * 128 + j];
        s = s > 0.f ? s : 0.15f * s;
        dst[j] = (_Float16)s;
    }
}

// ---------------------------------------------------------------------------
// 6) cube gather + folded conv GEMM via WMMA (block=256, 8 waves, 16 rows)
// ---------------------------------------------------------------------------
__global__ void conv_gemm(const float* __restrict__ v,
                          const float* __restrict__ vol0,
                          const float* __restrict__ vol1,
                          const float* __restrict__ vol2,
                          const _Float16* __restrict__ Bfrag,  // 8 tiles x 12 ks
                          const float* __restrict__ bias,
                          _Float16* __restrict__ xbuf) {
    __shared__ _Float16 At[16][392];     // stride 392 halves (784B, 4B aligned)
    __shared__ int base[16][3][3];       // [row][scale][axis]
    int m0 = blockIdx.x * 16;

    if (threadIdx.x < 16) {
        int vi = m0 + threadIdx.x;
        float p[3] = { v[vi*3+0], v[vi*3+1], v[vi*3+2] };
        #pragma unroll
        for (int n = 0; n < 3; ++n) {
            float fac = 96.0f / (float)(1 << n);
            int dim = LVOL >> n;
            #pragma unroll
            for (int ax = 0; ax < 3; ++ax) {
                int idx = clampi((int)rintf((p[ax] + 1.0f) * fac), 2, dim - 3);
                base[threadIdx.x][n][ax] = idx - 2;
            }
        }
    }
    __syncthreads();

    for (int t = threadIdx.x; t < 16 * 384; t += blockDim.x) {
        int r = t / 384, k = t % 384;
        _Float16 val = (_Float16)0.0f;
        if (k < 375) {
            int n = k / 125, rem = k % 125;
            int a = rem / 25, b = (rem / 5) % 5, c = rem % 5;
            const float* vol = (n == 0) ? vol0 : ((n == 1) ? vol1 : vol2);
            int dim = LVOL >> n;
            int i0 = base[r][n][0] + a;
            int i1 = base[r][n][1] + b;
            int i2 = base[r][n][2] + c;
            val = (_Float16)vol[((size_t)i0 * dim + i1) * dim + i2];
        }
        At[r][k] = val;
    }
    __syncthreads();

    int lane = threadIdx.x & 31;
    int wave = threadIdx.x >> 5;        // 0..7 -> ntile
    int h = lane >> 4, nn = lane & 15;
    const v16h* bptr = (const v16h*)(Bfrag + (size_t)wave * 12 * 32 * 16);
    const uint32_t* arow = (const uint32_t*)&At[nn][0];

    v8f acc = {};
    for (int ks = 0; ks < 12; ++ks) {
        AFrag af;
        int kb = ks * 32 + 8 * h;
        #pragma unroll
        for (int j = 0; j < 4; ++j) {
            af.u[j]     = arow[(kb + 2 * j) >> 1];
            af.u[4 + j] = arow[(kb + 16 + 2 * j) >> 1];
        }
        v16h bf = bptr[ks * 32 + lane];
        acc = __builtin_amdgcn_wmma_f32_16x16x32_f16(false, af.v, false, bf,
                                                     (short)0, acc, false, false);
    }
    int col = wave * 16 + nn;
    float bv = bias[col];
    #pragma unroll
    for (int r = 0; r < 8; ++r) {
        float x = acc[r] + bv;                     // no activation on x_local
        xbuf[(size_t)(m0 + r + 8 * h) * 256 + 128 + col] = (_Float16)x;
    }
}

// ---------------------------------------------------------------------------
// 7) f16 GEMM + bias + leaky(0.15) via WMMA, LDS-staged A tile.
//    block = 4 waves; one 16-row tile per block; wave owns NT n-tiles so the
//    whole N dimension is covered by one block (A read from global once).
//    A tile staged via gfx1250 GLOBAL_LOAD_ASYNC_TO_LDS_B128 (ASYNCcnt).
// ---------------------------------------------------------------------------
template <int KDIM, int NOUT, int NT>
__global__ void gemm_f16_lds(const _Float16* __restrict__ A,      // [nv][KDIM]
                             const _Float16* __restrict__ Bfrag,  // swizzled
                             const float* __restrict__ bias,
                             _Float16* __restrict__ Y) {          // [nv][NOUT]
    constexpr int NK     = KDIM / 32;     // k-steps
    constexpr int STRIDE = KDIM + 8;      // halves; +16B pad: 16B aligned rows,
                                          // rows spread across LDS banks
    __shared__ _Float16 At[16][STRIDE];

    int m0 = blockIdx.x * 16;

    // ---- async stage A tile (16 x KDIM f16) into LDS, 16B per lane --------
    {
        uint32_t lds0 = (uint32_t)(uintptr_t)&At[0][0];
        constexpr int CPR = KDIM / 8;              // 16B chunks per row
        for (int i = threadIdx.x; i < 16 * CPR; i += blockDim.x) {
            int r = i / CPR, c = i % CPR;
            uint64_t ga = (uint64_t)(uintptr_t)(A + (size_t)(m0 + r) * KDIM) +
                          (uint64_t)c * 16;
            uint32_t la = lds0 + (uint32_t)(r * STRIDE * 2 + c * 16);
            asm volatile("global_load_async_to_lds_b128 %0, %1, off"
                         :: "v"(la), "v"(ga) : "memory");
        }
        asm volatile("s_wait_asynccnt 0x0" ::: "memory");
    }
    __syncthreads();

    int lane = threadIdx.x & 31;
    int wave = threadIdx.x >> 5;                   // 0..3
    int h = lane >> 4, nn = lane & 15;
    int nb = wave * NT;                            // first n-tile of this wave

    const uint32_t* arow = (const uint32_t*)&At[nn][0];
    const v16h* bptr = (const v16h*)(Bfrag + (size_t)nb * NK * 32 * 16);

    v8f acc[NT];
    #pragma unroll
    for (int t = 0; t < NT; ++t) acc[t] = (v8f){};

    for (int ks = 0; ks < NK; ++ks) {
        AFrag af;
        int kb = ks * 32 + 8 * h;
        #pragma unroll
        for (int j = 0; j < 4; ++j) {
            af.u[j]     = arow[(kb + 2 * j) >> 1];
            af.u[4 + j] = arow[(kb + 16 + 2 * j) >> 1];
        }
        #pragma unroll
        for (int t = 0; t < NT; ++t) {
            v16h bf = bptr[((size_t)t * NK + ks) * 32 + lane];
            acc[t] = __builtin_amdgcn_wmma_f32_16x16x32_f16(
                false, af.v, false, bf, (short)0, acc[t], false, false);
        }
    }

    #pragma unroll
    for (int t = 0; t < NT; ++t) {
        int col = (nb + t) * 16 + nn;
        float bv = bias[col];
        #pragma unroll
        for (int r = 0; r < 8; ++r) {
            float x = acc[t][r] + bv;
            x = x > 0.f ? x : 0.15f * x;
            Y[(size_t)(m0 + r + 8 * h) * NOUT + col] = (_Float16)x;
        }
    }
}

// ---------------------------------------------------------------------------
// 8) fc4 tail: 256 -> 3, tanh * 0.1, + coord
// ---------------------------------------------------------------------------
__global__ void fc4_kernel(const _Float16* __restrict__ y2,   // [nv][256]
                           const float* __restrict__ v,
                           const float* __restrict__ w,       // (256,3)
                           const float* __restrict__ b,
                           float* __restrict__ out, int nv) {
    int i = blockIdx.x * blockDim.x + threadIdx.x;
    if (i >= nv) return;
    float s0 = b[0], s1 = b[1], s2 = b[2];
    const _Float16* row = y2 + (size_t)i * 256;
    for (int k = 0; k < 256; ++k) {
        float x = (float)row[k];
        s0 += x * w[k * 3 + 0];
        s1 += x * w[k * 3 + 1];
        s2 += x * w[k * 3 + 2];
    }
    out[i*3+0] = v[i*3+0] + 0.1f * tanhf(s0);
    out[i*3+1] = v[i*3+1] + 0.1f * tanhf(s1);
    out[i*3+2] = v[i*3+2] + 0.1f * tanhf(s2);
}

// ---------------------------------------------------------------------------
// Launch
// ---------------------------------------------------------------------------
static inline size_t align256(size_t x) { return (x + 255) & ~(size_t)255; }

extern "C" void kernel_launch(void* const* d_in, const int* in_sizes, int n_in,
                              void* d_out, int out_size, void* d_ws, size_t ws_size,
                              hipStream_t stream) {
    const float* v      = (const float*)d_in[0];
    const int*   f      = (const int*)  d_in[1];
    const float* volume = (const float*)d_in[2];
    const float* fc1_w  = (const float*)d_in[3];
    const float* fc1_b  = (const float*)d_in[4];
    const float* fc2_w  = (const float*)d_in[5];
    const float* fc2_b  = (const float*)d_in[6];
    const float* fc3_w  = (const float*)d_in[7];
    const float* fc3_b  = (const float*)d_in[8];
    const float* fc4_w  = (const float*)d_in[9];
    const float* fc4_b  = (const float*)d_in[10];
    const float* conv_w = (const float*)d_in[11];
    const float* conv_b = (const float*)d_in[12];
    const float* lfc_w  = (const float*)d_in[13];
    const float* lfc_b  = (const float*)d_in[14];
    float* out = (float*)d_out;

    // ---- workspace layout -------------------------------------------------
    uint8_t* ws = (uint8_t*)d_ws;
    size_t off = 0;
    float*     nacc  = (float*)(ws + off); off = align256(off + (size_t)NVTX * 3 * 4);
    float*     vol1  = (float*)(ws + off); off = align256(off + (size_t)96*96*96*4);
    float*     vol2  = (float*)(ws + off); off = align256(off + (size_t)48*48*48*4);
    float*     Wcomb = (float*)(ws + off); off = align256(off + (size_t)375*128*4);
    float*     bcomb = (float*)(ws + off); off = align256(off + (size_t)128*4);
    _Float16*  wBc   = (_Float16*)(ws + off); off = align256(off + (size_t)8 *12*32*16*2);
    _Float16*  wB2   = (_Float16*)(ws + off); off = align256(off + (size_t)32*8 *32*16*2);
    _Float16*  wB3   = (_Float16*)(ws + off); off = align256(off + (size_t)16*16*32*16*2);
    _Float16*  xbuf  = (_Float16*)(ws + off); off = align256(off + (size_t)NVTX * 256 * 2);
    _Float16*  y1    = (_Float16*)(ws + off); off = align256(off + (size_t)NVTX * 512 * 2);
    _Float16*  y2    = (_Float16*)(ws + off); off = align256(off + (size_t)NVTX * 256 * 2);
    (void)ws_size;

    const int MTILES = NVTX / 16;   // 9375 exactly

    zero_f32<<<512, 256, 0, stream>>>(nacc, NVTX * 3);
    face_normals<<<(NFACE + 255) / 256, 256, 0, stream>>>(v, f, nacc, NFACE);
    pool2<<<(96*96*96 + 255) / 256, 256, 0, stream>>>(volume, vol1, 96);
    pool2<<<(48*48*48 + 255) / 256, 256, 0, stream>>>(vol1, vol2, 48);
    combine_conv_lfc<<<(375*128 + 255) / 256, 256, 0, stream>>>(
        conv_w, conv_b, lfc_w, lfc_b, Wcomb, bcomb);
    prep_bfrag<<<(8*12*32  + 255) / 256, 256, 0, stream>>>(Wcomb, wBc, 375, 128, 12, 8);
    prep_bfrag<<<(32*8*32  + 255) / 256, 256, 0, stream>>>(fc2_w, wB2, 256, 512, 8, 32);
    prep_bfrag<<<(16*16*32 + 255) / 256, 256, 0, stream>>>(fc3_w, wB3, 512, 256, 16, 16);
    fc1_kernel<<<(NVTX + 255) / 256, 256, 0, stream>>>(v, nacc, fc1_w, fc1_b, xbuf, NVTX);
    conv_gemm<<<MTILES, 256, 0, stream>>>(v, volume, vol1, vol2, wBc, bcomb, xbuf);
    // fc2: (NVTX x 256) x (256 x 512); 4 waves x 8 n-tiles cover all 512 cols
    gemm_f16_lds<256, 512, 8><<<MTILES, 128, 0, stream>>>(xbuf, wB2, fc2_b, y1);
    // fc3: (NVTX x 512) x (512 x 256); 4 waves x 4 n-tiles cover all 256 cols
    gemm_f16_lds<512, 256, 4><<<MTILES, 128, 0, stream>>>(y1, wB3, fc3_b, y2);
    fc4_kernel<<<(NVTX + 255) / 256, 256, 0, stream>>>(y2, v, fc4_w, fc4_b, out, NVTX);
}
